// WSFAttention_63110249447961
// MI455X (gfx1250) — compile-verified, matrix operand order checked by reading, not run on hardware
//
#include <hip/hip_runtime.h>
#include <hip/hip_bf16.h>
#include <stdint.h>

// ---------------------------------------------------------------------------
// WSFAttention for MI455X (gfx1250): bf16 WMMA NT-GEMM pipeline with
// double-buffered GLOBAL_LOAD_ASYNC_TO_LDS staging (ASYNCcnt) in the big GEMM.
//   C=384, P=64, B=8, L=4096, n=1024, NH=3, d=384, SCALE=1/sqrt(128)
// ---------------------------------------------------------------------------

typedef __attribute__((ext_vector_type(16))) __bf16 v16bf;
typedef __attribute__((ext_vector_type(8)))  __bf16 v8bf;
typedef __attribute__((ext_vector_type(8)))  float  v8f;
typedef __attribute__((ext_vector_type(4)))  int    v4i;

#define CDIM   384
#define BATCH  8
#define LTOK   4096
#define NTOK   1024
#define SCALE2 0.08838834764831845f   // (384/3)^-0.5 = 128^-0.5
#define LDSP   40                     // LDS row pitch (32 K-elems + 8 pad) -> conflict-free

// ---------------- CDNA5 async copy (global -> LDS, 16B per lane) ------------
#if defined(__HIP_DEVICE_COMPILE__) && __has_builtin(__builtin_amdgcn_global_load_async_to_lds_b128)
#define HAVE_ASYNC_BUILTIN 1
#else
#define HAVE_ASYNC_BUILTIN 0
#endif

typedef __attribute__((address_space(1))) v4i gv4i_t;   // global int4
typedef __attribute__((address_space(3))) v4i lv4i_t;   // LDS int4

__device__ __forceinline__ void async_copy16(const void* g, void* l) {
#if HAVE_ASYNC_BUILTIN
  __builtin_amdgcn_global_load_async_to_lds_b128(
      (gv4i_t*)(uintptr_t)g, (lv4i_t*)(uint32_t)(uintptr_t)l, 0, 0);
#else
  asm volatile("global_load_async_to_lds_b128 %0, %1, off"
               :: "v"((uint32_t)(uintptr_t)l), "v"((uint64_t)(uintptr_t)g)
               : "memory");
#endif
}

__device__ __forceinline__ void wait_async4() {
#if defined(__HIP_DEVICE_COMPILE__) && __has_builtin(__builtin_amdgcn_s_wait_asynccnt)
  __builtin_amdgcn_s_wait_asynccnt(4);
#else
  asm volatile("s_wait_asynccnt 0x4" ::: "memory");
#endif
}
__device__ __forceinline__ void wait_async0() {
#if defined(__HIP_DEVICE_COMPILE__) && __has_builtin(__builtin_amdgcn_s_wait_asynccnt)
  __builtin_amdgcn_s_wait_asynccnt(0);
#else
  asm volatile("s_wait_asynccnt 0x0" ::: "memory");
#endif
}

// Build one 16x32 bf16 WMMA operand fragment from an LDS row block.
// Per CDNA5 ISA 16-bit operand layout: lanes 0-15 hold K=0..7 & 16..23,
// lanes 16-31 hold K=8..15 & 24..31 (element i -> VGPR i/2, half i%2).
__device__ __forceinline__ v16bf lds_frag(const __bf16* base, int sel) {
  v8bf lo = *(const v8bf*)(base + sel * 8);
  v8bf hi = *(const v8bf*)(base + 16 + sel * 8);
  v16bf o;
#pragma unroll
  for (int i = 0; i < 8; ++i) { o[i] = lo[i]; o[i + 8] = hi[i]; }
  return o;
}

// Direct-from-global fragment (used by the small-K projection GEMM).
__device__ __forceinline__ v16bf load_frag_nt(const __bf16* __restrict__ p,
                                              int ld, int rc, int k0, int lane) {
  const int r   = rc + (lane & 15);
  const int sel = (lane >> 4) & 1;
  const __bf16* base = p + (size_t)r * ld + k0 + sel * 8;
  v8bf lo = *(const v8bf*)(base);
  v8bf hi = *(const v8bf*)(base + 16);
  v16bf o;
#pragma unroll
  for (int i = 0; i < 8; ++i) { o[i] = lo[i]; o[i + 8] = hi[i]; }
  return o;
}

// ---------------------------------------------------------------------------
// f32 -> bf16 conversion
// ---------------------------------------------------------------------------
__global__ void cvt_bf16_kernel(const float* __restrict__ src,
                                __bf16* __restrict__ dst, int nelem) {
  int i = blockIdx.x * 256 + threadIdx.x;
  if (i < nelem) dst[i] = (__bf16)src[i];
}

// ---------------------------------------------------------------------------
// Projection GEMM + bias + LayerNorm (row width N = NT*128).
// Block = 16 rows of A, full N columns. 8 waves, each owns NT column tiles.
// ---------------------------------------------------------------------------
template <int NT>
__global__ __launch_bounds__(256)
void gemm_ln_kernel(const __bf16* __restrict__ A, const __bf16* __restrict__ W,
                    const float* __restrict__ bias, const float* __restrict__ lnw,
                    const float* __restrict__ lnb, __bf16* __restrict__ out, int K) {
  constexpr int N = NT * 16 * 8;
  __shared__ float smem[16 * N];
  const int lane = threadIdx.x & 31;
  const int wave = threadIdx.x >> 5;
  const int m0   = blockIdx.x * 16;
  const int n0   = wave * NT * 16;

  v8f acc[NT];
#pragma unroll
  for (int t = 0; t < NT; ++t) acc[t] = (v8f)(0.0f);

  for (int k0 = 0; k0 < K; k0 += 32) {
    v16bf a = load_frag_nt(A, K, m0, k0, lane);
#pragma unroll
    for (int t = 0; t < NT; ++t) {
      v16bf b = load_frag_nt(W, K, n0 + t * 16, k0, lane);
      acc[t] = __builtin_amdgcn_wmma_f32_16x16x32_bf16(
          false, a, false, b, (short)0, acc[t], false, false);
    }
  }

  const int sel = (lane >> 4) & 1, l15 = lane & 15;
#pragma unroll
  for (int t = 0; t < NT; ++t) {
    const int cn = n0 + t * 16 + l15;
    const float bv = bias[cn];
#pragma unroll
    for (int v = 0; v < 8; ++v)
      smem[(v + sel * 8) * N + cn] = acc[t][v] + bv;
  }
  __syncthreads();

  const int row = threadIdx.x >> 4;
  const int chunk = threadIdx.x & 15;
  const int CW = N / 16;
  const float* rp = &smem[row * N + chunk * CW];
  float s = 0.f, ss = 0.f;
  for (int i = 0; i < CW; ++i) { float x = rp[i]; s += x; ss += x * x; }
#pragma unroll
  for (int msk = 1; msk < 16; msk <<= 1) {
    s  += __shfl_xor(s,  msk);
    ss += __shfl_xor(ss, msk);
  }
  const float mean = s / (float)N;
  const float var  = ss / (float)N - mean * mean;
  const float rstd = rsqrtf(var + 1e-5f);
  __bf16* orow = out + (size_t)(m0 + row) * N;
  for (int i = 0; i < CW; ++i) {
    int cidx = chunk * CW + i;
    float x = rp[i];
    orow[cidx] = (__bf16)((x - mean) * rstd * lnw[cidx] + lnb[cidx]);
  }
}

// ---------------------------------------------------------------------------
// General batched NT GEMM with async-LDS double buffering.
//   C[z,m,n] = sum_k A[z,m,k] * B[z,n,k] (+bias[n])
// Block tile 128x128, K-slab 32. 8 waves as 4(m) x 2(n), wave = 32x64.
// Per slab each thread issues 4 GLOBAL_LOAD_ASYNC_TO_LDS_B128 (A+B), giving
// 4 ASYNCcnt ticks/wave; s_wait_asynccnt(4) retires the previous slab while
// the next streams in. Fragments then come from LDS via ds_load_b128.
// ---------------------------------------------------------------------------
__global__ __launch_bounds__(256)
void gemm_nt_kernel(const __bf16* __restrict__ A, const __bf16* __restrict__ B,
                    float* __restrict__ Cf, __bf16* __restrict__ Ch,
                    const float* __restrict__ bias,
                    int K, int lda, int ldb, int ldc,
                    long long sA, long long sB, long long sC) {
  __shared__ __bf16 As[2][128 * LDSP];
  __shared__ __bf16 Bs[2][128 * LDSP];
  const int z = blockIdx.z;
  A += (size_t)z * sA;
  B += (size_t)z * sB;
  const int lane = threadIdx.x & 31;
  const int wave = threadIdx.x >> 5;
  const int wm = wave >> 1, wn = wave & 1;
  const int m0 = blockIdx.x * 128;
  const int n0 = blockIdx.y * 128;

  const int lrow   = threadIdx.x >> 1;   // 0..127
  const int lchunk = threadIdx.x & 1;    // 16-element half of the 32-K slab

  auto stage = [&](int buf, int k0) {
    const __bf16* ga = A + (size_t)(m0 + lrow) * lda + k0 + lchunk * 16;
    const __bf16* gb = B + (size_t)(n0 + lrow) * ldb + k0 + lchunk * 16;
    __bf16* la = &As[buf][lrow * LDSP + lchunk * 16];
    __bf16* lb = &Bs[buf][lrow * LDSP + lchunk * 16];
    async_copy16(ga,     la);
    async_copy16(ga + 8, la + 8);
    async_copy16(gb,     lb);
    async_copy16(gb + 8, lb + 8);
  };

  v8f acc[2][4];
#pragma unroll
  for (int i = 0; i < 2; ++i)
#pragma unroll
    for (int j = 0; j < 4; ++j) acc[i][j] = (v8f)(0.0f);

  stage(0, 0);
  const int nk = K / 32;
  const int sel = (lane >> 4) & 1;
  const int fr  = lane & 15;

  for (int ks = 0; ks < nk; ++ks) {
    const int buf = ks & 1;
    if (ks + 1 < nk) { stage(buf ^ 1, (ks + 1) * 32); wait_async4(); }
    else             { wait_async0(); }
    __syncthreads();   // tile published to all waves

    v16bf a0 = lds_frag(&As[buf][(wm * 32 +      fr) * LDSP], sel);
    v16bf a1 = lds_frag(&As[buf][(wm * 32 + 16 + fr) * LDSP], sel);
#pragma unroll
    for (int j = 0; j < 4; ++j) {
      v16bf b = lds_frag(&Bs[buf][(wn * 64 + j * 16 + fr) * LDSP], sel);
      acc[0][j] = __builtin_amdgcn_wmma_f32_16x16x32_bf16(
          false, a0, false, b, (short)0, acc[0][j], false, false);
      acc[1][j] = __builtin_amdgcn_wmma_f32_16x16x32_bf16(
          false, a1, false, b, (short)0, acc[1][j], false, false);
    }
    __syncthreads();   // all waves done reading buf before it is overwritten
  }

#pragma unroll
  for (int i = 0; i < 2; ++i) {
#pragma unroll
    for (int j = 0; j < 4; ++j) {
      const int cn = n0 + wn * 64 + j * 16 + fr;
      const float bv = bias ? bias[cn] : 0.0f;
#pragma unroll
      for (int v = 0; v < 8; ++v) {
        const int rm = m0 + wm * 32 + i * 16 + v + sel * 8;
        const float val = acc[i][j][v] + bv;
        if (Cf) Cf[(size_t)z * sC + (size_t)rm * ldc + cn] = val;
        else    Ch[(size_t)z * sC + (size_t)rm * ldc + cn] = (__bf16)val;
      }
    }
  }
}

// ---------------------------------------------------------------------------
// Row softmax over 1024 elements; optional transposed write (produces P2^T).
// ---------------------------------------------------------------------------
__global__ __launch_bounds__(256)
void softmax_kernel(const float* __restrict__ S, __bf16* __restrict__ P,
                    float scale, int transposed) {
  __shared__ float red[8];
  const int row = blockIdx.x;
  const int mat = row >> 10;
  const int i   = row & 1023;
  const float* srow = S + (size_t)row * 1024;
  const int lane = threadIdx.x & 31;
  const int wave = threadIdx.x >> 5;

  float v[4];
  float mx = -3.4e38f;
#pragma unroll
  for (int jj = 0; jj < 4; ++jj) {
    v[jj] = srow[threadIdx.x + 256 * jj] * scale;
    mx = fmaxf(mx, v[jj]);
  }
#pragma unroll
  for (int msk = 16; msk >= 1; msk >>= 1) mx = fmaxf(mx, __shfl_xor(mx, msk));
  if (lane == 0) red[wave] = mx;
  __syncthreads();
  float m = red[0];
#pragma unroll
  for (int w = 1; w < 8; ++w) m = fmaxf(m, red[w]);
  __syncthreads();

  float sum = 0.f;
#pragma unroll
  for (int jj = 0; jj < 4; ++jj) { v[jj] = __expf(v[jj] - m); sum += v[jj]; }
#pragma unroll
  for (int msk = 16; msk >= 1; msk >>= 1) sum += __shfl_xor(sum, msk);
  if (lane == 0) red[wave] = sum;
  __syncthreads();
  float tot = 0.f;
#pragma unroll
  for (int w = 0; w < 8; ++w) tot += red[w];
  const float inv = 1.0f / tot;

  __bf16* pb = P + (size_t)mat * 1024 * 1024;
#pragma unroll
  for (int jj = 0; jj < 4; ++jj) {
    const int j = threadIdx.x + 256 * jj;
    const float p = v[jj] * inv;
    if (transposed) pb[(size_t)j * 1024 + i] = (__bf16)p;
    else            pb[(size_t)i * 1024 + j] = (__bf16)p;
  }
}

// ---------------------------------------------------------------------------
// DWT scatter: build Q1,Q2,K1,K2,V1^T,V2^T from t_sf (C ch) / t_wf (2C ch).
//   v1=ee v2=oe v3=eo v4=oo (each /2):
//   s0=.5( v1+v2+v3+v4) s1=.5(-v1-v2+v3+v4) s2=.5(-v1+v2-v3+v4) s3=.5(v1-v2-v3+v4)
// ---------------------------------------------------------------------------
__global__ void dwt_kernel(const __bf16* __restrict__ tsf, const __bf16* __restrict__ twf,
                           __bf16* __restrict__ Q1, __bf16* __restrict__ Q2,
                           __bf16* __restrict__ K1, __bf16* __restrict__ K2,
                           __bf16* __restrict__ V1t, __bf16* __restrict__ V2t) {
  const int idx = blockIdx.x * 256 + threadIdx.x;
  if (idx >= BATCH * NTOK * 1152) return;
  const int c2 = idx % 1152;
  const int m  = (idx / 1152) & 1023;
  const int b  = idx / (1152 * 1024);
  const int yy = m >> 5, xx = m & 31;
  const int r  = b * 4096 + (yy * 2) * 64 + xx * 2;

  if (c2 < CDIM) {
    const float v1 = (float)tsf[(size_t)(r)      * CDIM + c2];
    const float v2 = (float)tsf[(size_t)(r + 64) * CDIM + c2];
    const float v3 = (float)tsf[(size_t)(r + 1)  * CDIM + c2];
    const float v4 = (float)tsf[(size_t)(r + 65) * CDIM + c2];
    const float s0 = 0.5f * ( v1 + v2 + v3 + v4);
    const float s1 = 0.5f * (-v1 - v2 + v3 + v4);
    const float s2 = 0.5f * (-v1 + v2 - v3 + v4);
    const float s3 = 0.5f * ( v1 - v2 - v3 + v4);
    Q1[((size_t)b * NTOK + m) * CDIM + c2]           = (__bf16)s0;
    Q2[(((size_t)b * 3 + 0) * NTOK + m) * CDIM + c2] = (__bf16)s1;
    Q2[(((size_t)b * 3 + 1) * NTOK + m) * CDIM + c2] = (__bf16)s2;
    Q2[(((size_t)b * 3 + 2) * NTOK + m) * CDIM + c2] = (__bf16)s3;
  } else {
    const int cw = c2 - CDIM;
    const float v1 = (float)twf[(size_t)(r)      * 768 + cw];
    const float v2 = (float)twf[(size_t)(r + 64) * 768 + cw];
    const float v3 = (float)twf[(size_t)(r + 1)  * 768 + cw];
    const float v4 = (float)twf[(size_t)(r + 65) * 768 + cw];
    const float s0 = 0.5f * ( v1 + v2 + v3 + v4);
    const float s1 = 0.5f * (-v1 - v2 + v3 + v4);
    const float s2 = 0.5f * (-v1 + v2 - v3 + v4);
    const float s3 = 0.5f * ( v1 - v2 - v3 + v4);
    if (cw < CDIM) {
      K1[((size_t)b * NTOK + m) * CDIM + cw]            = (__bf16)s0;
      K2[(((size_t)b * 3 + 1) * NTOK + m) * CDIM + cw]  = (__bf16)s1;
      V1t[((size_t)b * CDIM + cw) * NTOK + m]           = (__bf16)s2;
      V2t[(((size_t)b * 3 + 1) * CDIM + cw) * NTOK + m] = (__bf16)s3;
    } else {
      const int cc = cw - CDIM;
      K2[(((size_t)b * 3 + 0) * NTOK + m) * CDIM + cc]  = (__bf16)s0;
      K2[(((size_t)b * 3 + 2) * NTOK + m) * CDIM + cc]  = (__bf16)s1;
      V2t[(((size_t)b * 3 + 0) * CDIM + cc) * NTOK + m] = (__bf16)s2;
      V2t[(((size_t)b * 3 + 2) * CDIM + cc) * NTOK + m] = (__bf16)s3;
    }
  }
}

// ---------------------------------------------------------------------------
// IWT: recombine [x1 | x2 heads] subbands into (B, L, C) bf16 for final GEMM.
// ---------------------------------------------------------------------------
__global__ void iwt_kernel(const __bf16* __restrict__ X1, const __bf16* __restrict__ X2t,
                           __bf16* __restrict__ xf) {
  const int idx = blockIdx.x * 256 + threadIdx.x;
  if (idx >= BATCH * LTOK * CDIM) return;
  const int c = idx % CDIM;
  const int t = idx / CDIM;
  const int x = t & 63, y = (t >> 6) & 63, b = t >> 12;
  const int m = (y >> 1) * 32 + (x >> 1);
  const float a0 = (float)X1[((size_t)b * NTOK + m) * CDIM + c];
  const float a1 = (float)X2t[(((size_t)b * 3 + 0) * NTOK + m) * CDIM + c];
  const float a2 = (float)X2t[(((size_t)b * 3 + 1) * NTOK + m) * CDIM + c];
  const float a3 = (float)X2t[(((size_t)b * 3 + 2) * NTOK + m) * CDIM + c];
  const int dy = y & 1, dx = x & 1;
  const float t1 = dx ? a1 : -a1;
  const float t2 = dy ? a2 : -a2;
  const float t3 = (dx == dy) ? a3 : -a3;
  xf[idx] = (__bf16)(0.5f * (a0 + t1 + t2 + t3));
}

// ---------------------------------------------------------------------------
// Host launcher
// ---------------------------------------------------------------------------
extern "C" void kernel_launch(void* const* d_in, const int* in_sizes, int n_in,
                              void* d_out, int out_size, void* d_ws, size_t ws_size,
                              hipStream_t stream) {
  (void)in_sizes; (void)n_in; (void)out_size; (void)ws_size;
  const float* sf     = (const float*)d_in[0];
  const float* wf     = (const float*)d_in[1];
  const float* q_w    = (const float*)d_in[2];
  const float* q_b    = (const float*)d_in[3];
  const float* q_lnw  = (const float*)d_in[4];
  const float* q_lnb  = (const float*)d_in[5];
  const float* kv_w   = (const float*)d_in[6];
  const float* kv_b   = (const float*)d_in[7];
  const float* kv_lnw = (const float*)d_in[8];
  const float* kv_lnb = (const float*)d_in[9];
  const float* out_w  = (const float*)d_in[10];
  const float* out_b  = (const float*)d_in[11];
  float* out = (float*)d_out;

  const size_t BL = (size_t)BATCH * LTOK;
  char* ws = (char*)d_ws;
  size_t off = 0;
  auto take = [&](size_t bytes) -> void* {
    void* p = ws + off;
    off = (off + bytes + 255) & ~(size_t)255;
    return p;
  };

  __bf16* sf_bf  = (__bf16*)take(BL * CDIM * 2);
  __bf16* wf_bf  = (__bf16*)take(BL * CDIM * 2);
  __bf16* qw_bf  = (__bf16*)take((size_t)CDIM * CDIM * 2);
  __bf16* kvw_bf = (__bf16*)take((size_t)2 * CDIM * CDIM * 2);
  __bf16* ow_bf  = (__bf16*)take((size_t)CDIM * CDIM * 2);
  __bf16* tsf    = (__bf16*)take(BL * CDIM * 2);
  __bf16* twf    = (__bf16*)take(BL * 2 * CDIM * 2);
  __bf16* Q1     = (__bf16*)take((size_t)BATCH * NTOK * CDIM * 2);
  __bf16* Q2     = (__bf16*)take((size_t)BATCH * 3 * NTOK * CDIM * 2);
  __bf16* K1     = (__bf16*)take((size_t)BATCH * NTOK * CDIM * 2);
  __bf16* K2     = (__bf16*)take((size_t)BATCH * 3 * NTOK * CDIM * 2);
  __bf16* V1t    = (__bf16*)take((size_t)BATCH * CDIM * NTOK * 2);
  __bf16* V2t    = (__bf16*)take((size_t)BATCH * 3 * CDIM * NTOK * 2);
  float*  S1     = (float*) take((size_t)BATCH * NTOK * NTOK * 4);
  float*  S2     = (float*) take((size_t)BATCH * 3 * NTOK * NTOK * 4);
  __bf16* P1     = (__bf16*)take((size_t)BATCH * NTOK * NTOK * 2);
  __bf16* P2t    = (__bf16*)take((size_t)BATCH * 3 * NTOK * NTOK * 2);
  __bf16* X1     = (__bf16*)take((size_t)BATCH * NTOK * CDIM * 2);
  __bf16* X2t    = (__bf16*)take((size_t)BATCH * 3 * NTOK * CDIM * 2);
  __bf16* xf     = (__bf16*)take(BL * CDIM * 2);

  auto cvt = [&](const float* s, __bf16* d, size_t n) {
    cvt_bf16_kernel<<<dim3((unsigned)((n + 255) / 256)), 256, 0, stream>>>(s, d, (int)n);
  };
  cvt(sf, sf_bf, BL * CDIM);
  cvt(wf, wf_bf, BL * CDIM);
  cvt(q_w, qw_bf, (size_t)CDIM * CDIM);
  cvt(kv_w, kvw_bf, (size_t)2 * CDIM * CDIM);
  cvt(out_w, ow_bf, (size_t)CDIM * CDIM);

  gemm_ln_kernel<3><<<dim3((unsigned)(BL / 16)), 256, 0, stream>>>(
      sf_bf, qw_bf, q_b, q_lnw, q_lnb, tsf, CDIM);
  gemm_ln_kernel<6><<<dim3((unsigned)(BL / 16)), 256, 0, stream>>>(
      wf_bf, kvw_bf, kv_b, kv_lnw, kv_lnb, twf, CDIM);

  dwt_kernel<<<dim3((BATCH * NTOK * 1152 + 255) / 256), 256, 0, stream>>>(
      tsf, twf, Q1, Q2, K1, K2, V1t, V2t);

  gemm_nt_kernel<<<dim3(8, 8, BATCH), 256, 0, stream>>>(
      Q1, K1, S1, nullptr, nullptr, CDIM, CDIM, CDIM, NTOK,
      (long long)NTOK * CDIM, (long long)NTOK * CDIM, (long long)NTOK * NTOK);
  gemm_nt_kernel<<<dim3(8, 8, BATCH * 3), 256, 0, stream>>>(
      Q2, K2, S2, nullptr, nullptr, CDIM, CDIM, CDIM, NTOK,
      (long long)NTOK * CDIM, (long long)NTOK * CDIM, (long long)NTOK * NTOK);

  softmax_kernel<<<dim3(BATCH * NTOK), 256, 0, stream>>>(S1, P1, 1.0f, 0);
  softmax_kernel<<<dim3(BATCH * 3 * NTOK), 256, 0, stream>>>(S2, P2t, SCALE2, 1);

  gemm_nt_kernel<<<dim3(8, 3, BATCH), 256, 0, stream>>>(
      P1, V1t, nullptr, X1, nullptr, NTOK, NTOK, NTOK, CDIM,
      (long long)NTOK * NTOK, (long long)CDIM * NTOK, (long long)NTOK * CDIM);
  gemm_nt_kernel<<<dim3(8, 3, BATCH * 3), 256, 0, stream>>>(
      P2t, V2t, nullptr, X2t, nullptr, NTOK, NTOK, NTOK, CDIM,
      (long long)NTOK * NTOK, (long long)CDIM * NTOK, (long long)NTOK * CDIM);

  iwt_kernel<<<dim3((unsigned)((BL * CDIM + 255) / 256)), 256, 0, stream>>>(X1, X2t, xf);

  gemm_nt_kernel<<<dim3((unsigned)(BL / 128), 3, 1), 256, 0, stream>>>(
      xf, ow_bf, out, nullptr, out_b, CDIM, CDIM, CDIM, CDIM,
      0LL, 0LL, 0LL);
}
